// HATLayer_13202729467973
// MI455X (gfx1250) — compile-verified, matrix operand order checked by reading, not run on hardware
//
#include <hip/hip_runtime.h>
#include <hip/hip_bf16.h>

#define N_NODES 50000
#define N_EDGES 10000
#define NNZ     250000
#define DIM     256
#define HID     512
#define MTILES  5           // 80 rows per block; divides 50000/16=3125 and 250000/16=15625
#define MROWS   (MTILES * 16)

typedef __bf16 bf16;
typedef __bf16 v16bf __attribute__((ext_vector_type(16)));
typedef __bf16 v8bf  __attribute__((ext_vector_type(8)));
typedef float  v8f   __attribute__((ext_vector_type(8)));

// ---------- helpers ----------
__device__ inline unsigned f2ord(float x) {
    unsigned u = __float_as_uint(x);
    return (u & 0x80000000u) ? ~u : (u | 0x80000000u);
}
__device__ inline float ord2f(unsigned u) {
    return (u & 0x80000000u) ? __uint_as_float(u & 0x7FFFFFFFu) : __uint_as_float(~u);
}
__device__ inline void atomAddF(float* p, float v) { unsafeAtomicAdd(p, v); }

// A-fragment (16x32 bf16) from an LDS row: lane holds row m=lane&15.
// lanes 0-15: K = {k0..k0+7, k0+16..k0+23}, lanes 16-31: +8 (folded into k0).
__device__ inline v16bf load_a_frag(const bf16* row, int k0) {
    union { v16bf v; v8bf h[2]; } u;
    u.h[0] = *(const v8bf*)(row + k0);
    u.h[1] = *(const v8bf*)(row + k0 + 16);
    return u.v;
}

// Shared GEMM+activation+reduce core for the fused attention MLPs.
// a_lds: [MROWS][LDS_STRIDE] bf16 staged A rows; wfrag: pre-swizzled B frags.
template <int KTILES, int LDS_STRIDE>
__device__ inline void mlp_core(const bf16* a_lds, const bf16* wfrag,
                                const float* b1, const float* w2, float* rowsum) {
    const int tid  = threadIdx.x;
    const int lane = tid & 31, wave = tid >> 5;
    const int half = lane >> 4, ln = lane & 15;
    const int koff = half * 8;
    const v16bf* wf = (const v16bf*)wfrag;

    float acc[MTILES][8] = {};
    #pragma unroll
    for (int ntb = 0; ntb < 4; ++ntb) {
        const int ntile = wave * 4 + ntb;
        v8f c[MTILES] = {};
        for (int kt = 0; kt < KTILES; ++kt) {
            v16bf b = wf[(kt * 32 + ntile) * 32 + lane];  // 1 load, reused 5x
            #pragma unroll
            for (int mt = 0; mt < MTILES; ++mt) {
                v16bf a = load_a_frag(a_lds + (mt * 16 + ln) * LDS_STRIDE,
                                      kt * 32 + koff);
                c[mt] = __builtin_amdgcn_wmma_f32_16x16x32_bf16(
                            false, a, false, b, (short)0, c[mt], false, false);
            }
        }
        const int col = ntile * 16 + ln;
        const float bb = b1[col], ww = w2[col];
        #pragma unroll
        for (int mt = 0; mt < MTILES; ++mt)
            #pragma unroll
            for (int r = 0; r < 8; ++r) {
                float t = c[mt][r] + bb;
                t = (t > 0.f) ? t : 0.2f * t;   // LeakyReLU(0.2)
                acc[mt][r] += t * ww;
            }
    }
    #pragma unroll
    for (int mt = 0; mt < MTILES; ++mt)
        #pragma unroll
        for (int r = 0; r < 8; ++r) {
            float a = acc[mt][r];
            a += __shfl_xor(a, 1);
            a += __shfl_xor(a, 2);
            a += __shfl_xor(a, 4);
            a += __shfl_xor(a, 8);
            if (ln == 0) atomicAdd(&rowsum[mt * 16 + r + 8 * half], a);
        }
}

// ---------- init ----------
__global__ void k_fill_u32(unsigned* p, unsigned val, int n) {
    int i = blockIdx.x * blockDim.x + threadIdx.x;
    if (i < n) p[i] = val;
}

__global__ void k_copy4(const float4* __restrict__ s, float4* __restrict__ d, int n) {
    int i = blockIdx.x * blockDim.x + threadIdx.x;
    if (i < n) d[i] = s[i];
}

// ---------- weight pre-swizzle into WMMA B-fragment order ----------
// src: (K x H) row-major.  dst: [ktile][ntile][lane][i] bf16, 512 per fragment.
__global__ void k_swz(const float* __restrict__ src, bf16* __restrict__ dst,
                      int H, int total) {
    int idx = blockIdx.x * blockDim.x + threadIdx.x;
    if (idx >= total) return;
    int i    = idx & 15;
    int lane = (idx >> 4) & 31;
    int frag = idx >> 9;
    int ktile = frag >> 5;      // ntiles = 32 (H=512)
    int ntile = frag & 31;
    int koff = (lane >= 16) ? 8 : 0;
    int k = ktile * 32 + ((i < 8) ? (koff + i) : (16 + koff + (i - 8)));
    int n = ntile * 16 + (lane & 15);
    dst[idx] = (bf16)src[k * H + n];
}

// ---------- fused MLP1: alpha1_node = (lrelu(Xv@W1+b1))@w2 + b2 ----------
#define S1 (DIM + 8)
__global__ __launch_bounds__(256) void k_mlp1(const float* __restrict__ Xv,
                                              const bf16* __restrict__ wfrag,
                                              const float* __restrict__ b1,
                                              const float* __restrict__ w2,
                                              const float* __restrict__ b2,
                                              float* __restrict__ out) {
    extern __shared__ __align__(16) char smem1[];
    bf16*  a_lds  = (bf16*)smem1;                        // [MROWS][S1]
    float* rowsum = (float*)(smem1 + MROWS * S1 * 2);    // [MROWS]
    const int tid  = threadIdx.x;
    const int base = blockIdx.x * MROWS;

    if (tid < MROWS) rowsum[tid] = 0.f;
    #pragma unroll
    for (int j = 0; j < MROWS; ++j) {       // 80*256 = MROWS*DIM elements
        int idx = tid + j * 256;
        int m = idx >> 8, k = idx & 255;
        a_lds[m * S1 + k] = (bf16)Xv[(base + m) * DIM + k];
    }
    __syncthreads();

    mlp_core<8, S1>(a_lds, wfrag, b1, w2, rowsum);

    __syncthreads();
    if (tid < MROWS) out[base + tid] = rowsum[tid] + b2[0];
}

// ---------- segment softmax passes ----------
__global__ void k_seg_max(const float* __restrict__ x, const int* __restrict__ gidx,
                          const int* __restrict__ seg, unsigned* __restrict__ mx, int n) {
    int i = blockIdx.x * blockDim.x + threadIdx.x;
    if (i >= n) return;
    float xi = gidx ? x[gidx[i]] : x[i];
    atomicMax(&mx[seg[i]], f2ord(xi));
}
__global__ void k_seg_exp(const float* __restrict__ x, const int* __restrict__ gidx,
                          const int* __restrict__ seg, const unsigned* __restrict__ mx,
                          float* __restrict__ sum, float* __restrict__ ex, int n) {
    int i = blockIdx.x * blockDim.x + threadIdx.x;
    if (i >= n) return;
    float xi = gidx ? x[gidx[i]] : x[i];
    float ev = __expf(xi - ord2f(mx[seg[i]]));
    ex[i] = ev;
    atomAddF(&sum[seg[i]], ev);
}
__global__ void k_seg_norm(float* __restrict__ ex, const int* __restrict__ seg,
                           const float* __restrict__ sum, int n) {
    int i = blockIdx.x * blockDim.x + threadIdx.x;
    if (i >= n) return;
    ex[i] = ex[i] / (sum[seg[i]] + 1e-16f);
}

// ---------- Xe accumulation: Xe[e] += Xv[v] * alpha1 ----------
__global__ __launch_bounds__(256) void k_xe(const float* __restrict__ Xv,
                                            const int* __restrict__ v,
                                            const int* __restrict__ e,
                                            const float* __restrict__ a1,
                                            float* __restrict__ Xe) {
    int nz = blockIdx.x, c = threadIdx.x;
    float a = a1[nz];
    atomAddF(&Xe[e[nz] * DIM + c], Xv[v[nz] * DIM + c] * a);
}

// ---------- LayerNorm on Xe (in place) ----------
__global__ __launch_bounds__(256) void k_ln(float* __restrict__ Xe,
                                            const float* __restrict__ g,
                                            const float* __restrict__ b) {
    __shared__ float red[256];
    int row = blockIdx.x, tid = threadIdx.x;
    float x = Xe[row * DIM + tid];
    red[tid] = x; __syncthreads();
    for (int s = 128; s > 0; s >>= 1) { if (tid < s) red[tid] += red[tid + s]; __syncthreads(); }
    float mu = red[0] * (1.f / DIM); __syncthreads();
    float d = x - mu;
    red[tid] = d * d; __syncthreads();
    for (int s = 128; s > 0; s >>= 1) { if (tid < s) red[tid] += red[tid + s]; __syncthreads(); }
    float var = red[0] * (1.f / DIM);
    Xe[row * DIM + tid] = d * rsqrtf(var + 1e-5f) * g[tid] + b[tid];
}

// ---------- fused MLP2: alpha2 logit over cat(Xe[e], Xv[v]*a1), K=512 ----------
#define S2 (2 * DIM + 8)
__global__ __launch_bounds__(256) void k_mlp2(const float* __restrict__ Xv,
                                              const float* __restrict__ Xe,
                                              const int* __restrict__ v,
                                              const int* __restrict__ e,
                                              const float* __restrict__ a1,
                                              const bf16* __restrict__ wfrag,
                                              const float* __restrict__ b1,
                                              const float* __restrict__ w2,
                                              const float* __restrict__ b2,
                                              float* __restrict__ out) {
    extern __shared__ __align__(16) char smem2[];
    bf16*  a_lds  = (bf16*)smem2;                        // [MROWS][S2]
    float* rowsum = (float*)(smem2 + MROWS * S2 * 2);    // [MROWS]
    int*   ei     = (int*)(rowsum + MROWS);              // [MROWS]
    int*   vi     = ei + MROWS;                          // [MROWS]
    float* a1s    = (float*)(vi + MROWS);                // [MROWS]
    const int tid  = threadIdx.x;
    const int base = blockIdx.x * MROWS;

    if (tid < MROWS) {
        ei[tid]  = e[base + tid];
        vi[tid]  = v[base + tid];
        a1s[tid] = a1[base + tid];
        rowsum[tid] = 0.f;
    }
    __syncthreads();

    #pragma unroll
    for (int j = 0; j < 2 * MROWS; ++j) {   // 160*256 = MROWS*2*DIM elements
        int idx = tid + j * 256;
        int m = idx >> 9, k = idx & 511;
        float val = (k < DIM) ? Xe[ei[m] * DIM + k]
                              : Xv[vi[m] * DIM + (k - DIM)] * a1s[m];
        a_lds[m * S2 + k] = (bf16)val;
    }
    __syncthreads();

    mlp_core<16, S2>(a_lds, wfrag, b1, w2, rowsum);

    __syncthreads();
    if (tid < MROWS) out[base + tid] = rowsum[tid] + b2[0];
}

// ---------- output scatter: out[v] += Xe[e] * alpha2  (out preloaded with Xv_in) ----------
__global__ __launch_bounds__(256) void k_scatter(const float* __restrict__ Xe,
                                                 const int* __restrict__ v,
                                                 const int* __restrict__ e,
                                                 const float* __restrict__ a2,
                                                 float* __restrict__ out) {
    int nz = blockIdx.x, c = threadIdx.x;
    float a = a2[nz];
    atomAddF(&out[v[nz] * DIM + c], Xe[e[nz] * DIM + c] * a);
}

// ---------- host ----------
extern "C" void kernel_launch(void* const* d_in, const int* in_sizes, int n_in,
                              void* d_out, int out_size, void* d_ws, size_t ws_size,
                              hipStream_t stream) {
    const float* Xv   = (const float*)d_in[0];
    const int*   v    = (const int*)d_in[1];
    const int*   e    = (const int*)d_in[2];
    const float* W1w1 = (const float*)d_in[3];
    const float* W1b1 = (const float*)d_in[4];
    const float* W1w2 = (const float*)d_in[5];
    const float* W1b2 = (const float*)d_in[6];
    const float* W2w1 = (const float*)d_in[7];
    const float* W2b1 = (const float*)d_in[8];
    const float* W2w2 = (const float*)d_in[9];
    const float* W2b2 = (const float*)d_in[10];
    const float* gam  = (const float*)d_in[11];
    const float* bet  = (const float*)d_in[12];
    float* out = (float*)d_out;

    char* p = (char*)d_ws;
    auto take = [&](size_t bytes) { void* r = (void*)p; p += (bytes + 255) & ~(size_t)255; return r; };
    // zeroed region (contiguous): emax, esum, nmax, nsum, Xe
    unsigned* emax = (unsigned*)take(N_EDGES * 4);
    float*    esum = (float*)take(N_EDGES * 4);
    unsigned* nmax = (unsigned*)take(N_NODES * 4);
    float*    nsum = (float*)take(N_NODES * 4);
    float*    Xe   = (float*)take((size_t)N_EDGES * DIM * 4);
    char* zero_end = p;
    float* a1_node = (float*)take(N_NODES * 4);
    float* a1      = (float*)take(NNZ * 4);
    float* a2      = (float*)take(NNZ * 4);
    bf16*  w1f     = (bf16*)take((size_t)8 * 32 * 512 * 2);
    bf16*  w2f     = (bf16*)take((size_t)16 * 32 * 512 * 2);

    // 0) zero-init accumulators
    int nz32 = (int)((zero_end - (char*)emax) / 4);
    k_fill_u32<<<(nz32 + 255) / 256, 256, 0, stream>>>(emax, 0u, nz32);

    // 1) pre-swizzle weights into WMMA fragment order
    k_swz<<<(8 * 32 * 512 + 255) / 256, 256, 0, stream>>>(W1w1, w1f, HID, 8 * 32 * 512);
    k_swz<<<(16 * 32 * 512 + 255) / 256, 256, 0, stream>>>(W2w1, w2f, HID, 16 * 32 * 512);

    // 2) node attention logits (80 rows/block, WMMA)
    size_t lds1 = (size_t)MROWS * S1 * 2 + MROWS * 4;
    k_mlp1<<<N_NODES / MROWS, 256, lds1, stream>>>(Xv, w1f, W1b1, W1w2, W1b2, a1_node);

    // 3) segment softmax over edges
    int gb = (NNZ + 255) / 256;
    k_seg_max<<<gb, 256, 0, stream>>>(a1_node, v, e, emax, NNZ);
    k_seg_exp<<<gb, 256, 0, stream>>>(a1_node, v, e, emax, esum, a1, NNZ);
    k_seg_norm<<<gb, 256, 0, stream>>>(a1, e, esum, NNZ);

    // 4) edge features + LayerNorm
    k_xe<<<NNZ, 256, 0, stream>>>(Xv, v, e, a1, Xe);
    k_ln<<<N_EDGES, 256, 0, stream>>>(Xe, gam, bet);

    // 5) pair attention logits (big WMMA GEMM, 80 rows/block)
    size_t lds2 = (size_t)MROWS * S2 * 2 + MROWS * 4 * 3 + MROWS * 4;
    k_mlp2<<<NNZ / MROWS, 256, lds2, stream>>>(Xv, Xe, v, e, a1, w2f, W2b1, W2w2, W2b2, a2);

    // 6) segment softmax over nodes
    k_seg_max<<<gb, 256, 0, stream>>>(a2, nullptr, v, nmax, NNZ);
    k_seg_exp<<<gb, 256, 0, stream>>>(a2, nullptr, v, nmax, nsum, a2, NNZ);
    k_seg_norm<<<gb, 256, 0, stream>>>(a2, v, nsum, NNZ);

    // 7) residual init + scatter output
    int n4 = N_NODES * DIM / 4;
    k_copy4<<<(n4 + 255) / 256, 256, 0, stream>>>((const float4*)Xv, (float4*)out, n4);
    k_scatter<<<NNZ, 256, 0, stream>>>(Xe, v, e, a2, out);
}
